// Decoder_77094662963487
// MI455X (gfx1250) — matrix operand order repaired and checked by reading, not compiled, not run on hardware
//
#include <hip/hip_runtime.h>
#include <hip/hip_bf16.h>

typedef __bf16 bf16_t;
typedef bf16_t v16bf  __attribute__((ext_vector_type(16)));
typedef bf16_t bf16x8 __attribute__((ext_vector_type(8)));
typedef bf16_t bf16x4 __attribute__((ext_vector_type(4)));
typedef float  v8f    __attribute__((ext_vector_type(8)));

#define NB      65536
#define LATENT  256

// workspace layout (bf16 element offsets)
#define WS_W1   0
#define WS_W2   (256*512)
#define WS_W3   (256*512 + 512*256)
#define WS_TOT  (256*512 + 512*256 + 256*32)

// LDS layout (bytes); regions reused across barrier-separated phases
#define OFF_Z   0          // 64x256 bf16 = 32768
#define OFF_H1  32768      // 64x512 bf16 = 65536
#define OFF_H2  0          // reuses z region: 64x256 bf16
#define OFF_P   32768      // reuses h1 region: 64x32 f32 = 8192
#define SMEM_BYTES 98304

union FragU { v16bf v; bf16x8 h[2]; };

// ---------------------------------------------------------------- weights->bf16
__global__ void convert_weights_kernel(const float* __restrict__ W1,
                                       const float* __restrict__ W2,
                                       const float* __restrict__ W3,
                                       bf16_t* __restrict__ ws) {
  int tid = blockIdx.x * 256 + threadIdx.x;
  if (tid < 256*512) {
    ws[WS_W1 + tid] = (bf16_t)W1[tid];
  } else if (tid < 256*512 + 512*256) {
    int t = tid - 256*512;
    ws[WS_W2 + t] = (bf16_t)W2[t];
  } else if (tid < WS_TOT) {
    int t = tid - (256*512 + 512*256);
    int k = t >> 5, n = t & 31;
    ws[WS_W3 + t] = (n < 31) ? (bf16_t)W3[k*31 + n] : (bf16_t)0.0f;
  }
}

// ---------------------------------------------------------------- WMMA GEMM tile
// A: 64xK bf16 in LDS (row-major, ldaBytes stride). B: KxN bf16 row-major in
// global (L2-resident weights). Each wave owns 16Mx64N tiles; per K-step the A
// fragment (2x ds_load_b128) is reused for 4 WMMAs against 4 B fragments
// (16 strided 2-byte global loads each).
template<int K, int N, bool RELU>
__device__ __forceinline__ void gemm_tile(
    const char* smemA, int ldaBytes,
    const bf16_t* __restrict__ Wb,
    const float* __restrict__ bias,
    char* smemOut, int ldoBytes,
    int wave, int lane)
{
  const int numWT = 4 * (N / 64);
  const int mlane = lane & 15;
  const int kb    = (lane >> 4) << 3;     // A chunk base per ISA 16-bit A layout
  for (int wt = wave; wt < numWT; wt += 8) {
    const int m16 = (wt & 3) << 4;
    const int n0  = (wt >> 2) << 6;
    v8f c[4] = {};
    const int Mrow = m16 + mlane;
    #pragma unroll
    for (int k0 = 0; k0 < K; k0 += 32) {
      const char* pa = smemA + Mrow * ldaBytes + (k0 + kb) * 2;
      FragU ua;
      ua.h[0] = *(const bf16x8*)(pa);        // K [kb, kb+8)
      ua.h[1] = *(const bf16x8*)(pa + 32);   // K [kb+16, kb+24)
      const bf16_t* pb = Wb + (size_t)(k0 + kb) * N + n0 + mlane; // col n0+mlane, K rows in table order
      #pragma unroll
      for (int j = 0; j < 4; ++j) {
        FragU ub;
        #pragma unroll
        for (int s = 0; s < 8; ++s) {
          ub.h[0][s] = pb[(size_t)s * N + j*16];
          ub.h[1][s] = pb[(size_t)(s + 16) * N + j*16];
        }
        c[j] = __builtin_amdgcn_wmma_f32_16x16x32_bf16(
                   false, ua.v, false, ub.v, (short)0, c[j], false, false);
      }
    }
    // epilogue: bias + relu + pack bf16 to LDS (C/D layout: VGPR r -> M=r+8*(lane/16), N=lane%16)
    const int rbase = m16 + ((lane >> 4) << 3);
    #pragma unroll
    for (int j = 0; j < 4; ++j) {
      const int col = n0 + (j << 4) + mlane;
      const float bv = bias[col];
      #pragma unroll
      for (int r = 0; r < 8; ++r) {
        float v = c[j][r] + bv;
        if (RELU) v = v > 0.0f ? v : 0.0f;
        *(bf16_t*)(smemOut + (rbase + r) * ldoBytes + col * 2) = (bf16_t)v;
      }
    }
  }
}

// ---------------------------------------------------------------- kinematics helpers
__device__ __forceinline__ void rodrigues3(float x, float y, float z, float R[9]) {
  float t2 = x*x + y*y + z*z + 1e-12f;
  float th = __builtin_sqrtf(t2);
  float inv = 1.0f / th;
  float kx = x * inv, ky = y * inv, kz = z * inv;
  float ct = __cosf(th), st = __sinf(th);
  float oc = 1.0f - ct;
  // R = ct*I + st*K + (1-ct)*k k^T   (K^2 = kk^T - I for unit k)
  R[0] = ct + oc*kx*kx;    R[1] = oc*kx*ky - st*kz; R[2] = oc*kx*kz + st*ky;
  R[3] = oc*kx*ky + st*kz; R[4] = ct + oc*ky*ky;    R[5] = oc*ky*kz - st*kx;
  R[6] = oc*kx*kz - st*ky; R[7] = oc*ky*kz + st*kx; R[8] = ct + oc*kz*kz;
}

__device__ __forceinline__ void joint_pos(const float* __restrict__ Jt,
                                          const float* __restrict__ sd,
                                          const float shape[10], int j, bool mirror,
                                          float Jout[3]) {
  #pragma unroll
  for (int k = 0; k < 3; ++k) {
    float v = Jt[j*3 + k];
    const float* s = sd + (j*3 + k) * 10;
    #pragma unroll
    for (int d = 0; d < 10; ++d) v += s[d] * shape[d];
    Jout[k] = v;
  }
  if (mirror) Jout[0] = -Jout[0];
}

// ---------------------------------------------------------------- fused decoder
__global__ __launch_bounds__(256)
void hand_decoder_kernel(const float* __restrict__ z_left,
                         const float* __restrict__ z_right,
                         const float* __restrict__ b1,
                         const float* __restrict__ b2,
                         const float* __restrict__ b3,
                         const float* __restrict__ Jt,
                         const float* __restrict__ shape_dirs,
                         const float* __restrict__ hc,
                         const float* __restrict__ hm,
                         const bf16_t* __restrict__ wbf,
                         float* __restrict__ out)
{
  extern __shared__ __align__(16) char smem[];
  const int tid  = threadIdx.x;
  const int wave = tid >> 5;
  const int lane = tid & 31;
  const int hand = blockIdx.x >> 10;     // 0=left, 1=right
  const int tile = blockIdx.x & 1023;
  const int row0 = tile * 64;
  const float* z = hand ? z_right : z_left;

  // ---- phase 0: stage 64x256 z tile into LDS as bf16 (float4 loads)
  #pragma unroll
  for (int it = 0; it < 16; ++it) {
    int idx = tid + it * 256;            // 4096 float4 total
    int r = idx >> 6, c4 = idx & 63;
    float4 f = *(const float4*)(z + (size_t)(row0 + r) * LATENT + c4 * 4);
    bf16x4 h;
    h.x = (bf16_t)f.x; h.y = (bf16_t)f.y; h.z = (bf16_t)f.z; h.w = (bf16_t)f.w;
    *(bf16x4*)(smem + OFF_Z + r * 512 + c4 * 8) = h;
  }
  __syncthreads();

  // ---- phase 1: h1 = relu(z @ W1 + b1)   (64x512, K=256)
  gemm_tile<256, 512, true>(smem + OFF_Z, 512, wbf + WS_W1, b1,
                            smem + OFF_H1, 1024, wave, lane);
  __syncthreads();

  // ---- phase 2: h2 = relu(h1 @ W2 + b2)  (64x256, K=512)
  gemm_tile<512, 256, true>(smem + OFF_H1, 1024, wbf + WS_W2, b2,
                            smem + OFF_H2, 512, wave, lane);
  __syncthreads();

  // ---- phase 3: p = h2 @ W3 + b3  (64x32, K=256), f32 out, no relu
  {
    const int m16 = (wave & 3) << 4;
    const int n0  = (wave >> 2) << 4;
    v8f c = {};
    const int Mrow = m16 + (lane & 15);
    const int kb   = (lane >> 4) << 3;
    const bf16_t* W3b = wbf + WS_W3;
    #pragma unroll
    for (int k0 = 0; k0 < 256; k0 += 32) {
      const char* pa = smem + OFF_H2 + Mrow * 512 + (k0 + kb) * 2;
      FragU ua;
      ua.h[0] = *(const bf16x8*)(pa);
      ua.h[1] = *(const bf16x8*)(pa + 32);
      const bf16_t* pb = W3b + (size_t)(k0 + kb) * 32 + n0 + (lane & 15);
      FragU ub;
      #pragma unroll
      for (int s = 0; s < 8; ++s) {
        ub.h[0][s] = pb[s * 32];
        ub.h[1][s] = pb[(s + 16) * 32];
      }
      c = __builtin_amdgcn_wmma_f32_16x16x32_bf16(
              false, ua.v, false, ub.v, (short)0, c, false, false);
    }
    const int col  = n0 + (lane & 15);
    const float bv = (col < 31) ? b3[col] : 0.0f;
    const int rbase = m16 + ((lane >> 4) << 3);
    #pragma unroll
    for (int r = 0; r < 8; ++r)
      *(float*)(smem + OFF_P + (rbase + r) * 128 + col * 4) = c[r] + bv;
  }
  __syncthreads();

  // ---- phase 4: per-row forward kinematics (threads 0..63, one row each)
  if (tid < 64) {
    const float* pr = (const float*)(smem + OFF_P + tid * 128);
    float pose[15], shape[10];
    #pragma unroll
    for (int i = 0; i < 15; ++i) pose[i] = pr[i];
    #pragma unroll
    for (int i = 0; i < 10; ++i) shape[i] = pr[15 + i];
    const float rx = pr[25], ry = pr[26], rz = pr[27];
    const float tx = pr[28], ty = pr[29], tz = pr[30];
    const bool mirror = (hand == 0);

    float* orow = out + (size_t)hand * (size_t)NB * 48 + (size_t)(row0 + tid) * 48;

    float J0[3];
    joint_pos(Jt, shape_dirs, shape, 0, mirror, J0);
    float Rroot[9];
    rodrigues3(rx, ry, rz, Rroot);
    orow[0] = J0[0] + tx; orow[1] = J0[1] + ty; orow[2] = J0[2] + tz;

    // PARENTS = 5 chains of 3 joints hanging off the root
    for (int ch = 0; ch < 5; ++ch) {
      float Rw[9], tw[3], Jp[3];
      #pragma unroll
      for (int i = 0; i < 9; ++i) Rw[i] = Rroot[i];
      tw[0] = J0[0]; tw[1] = J0[1]; tw[2] = J0[2];
      Jp[0] = J0[0]; Jp[1] = J0[1]; Jp[2] = J0[2];
      for (int s = 0; s < 3; ++s) {
        const int j = 1 + ch * 3 + s;
        // full_pose row for this joint: hm + pose @ hc
        float aa[3];
        #pragma unroll
        for (int k = 0; k < 3; ++k) {
          const int i = (j - 1) * 3 + k;
          float v = hm[i];
          #pragma unroll
          for (int d = 0; d < 15; ++d) v += pose[d] * hc[d * 45 + i];
          aa[k] = v;
        }
        float Rl[9];
        rodrigues3(aa[0], aa[1], aa[2], Rl);
        float Jj[3];
        joint_pos(Jt, shape_dirs, shape, j, mirror, Jj);
        // t_j = Rw_parent * (J_j - J_parent) + t_parent
        const float d0 = Jj[0] - Jp[0], d1 = Jj[1] - Jp[1], d2 = Jj[2] - Jp[2];
        float tn[3];
        tn[0] = Rw[0]*d0 + Rw[1]*d1 + Rw[2]*d2 + tw[0];
        tn[1] = Rw[3]*d0 + Rw[4]*d1 + Rw[5]*d2 + tw[1];
        tn[2] = Rw[6]*d0 + Rw[7]*d1 + Rw[8]*d2 + tw[2];
        // Rw_j = Rw_parent * R_j
        float Rn[9];
        #pragma unroll
        for (int rr = 0; rr < 3; ++rr)
          #pragma unroll
          for (int cc = 0; cc < 3; ++cc)
            Rn[rr*3+cc] = Rw[rr*3+0]*Rl[cc] + Rw[rr*3+1]*Rl[3+cc] + Rw[rr*3+2]*Rl[6+cc];
        #pragma unroll
        for (int i = 0; i < 9; ++i) Rw[i] = Rn[i];
        orow[j*3+0] = tn[0] + tx;
        orow[j*3+1] = tn[1] + ty;
        orow[j*3+2] = tn[2] + tz;
        tw[0] = tn[0]; tw[1] = tn[1]; tw[2] = tn[2];
        Jp[0] = Jj[0]; Jp[1] = Jj[1]; Jp[2] = Jj[2];
      }
    }
  }
}

// ---------------------------------------------------------------- launch
extern "C" void kernel_launch(void* const* d_in, const int* in_sizes, int n_in,
                              void* d_out, int out_size, void* d_ws, size_t ws_size,
                              hipStream_t stream) {
  (void)in_sizes; (void)n_in; (void)out_size; (void)ws_size;
  const float* z_left  = (const float*)d_in[0];
  const float* z_right = (const float*)d_in[1];
  const float* W1      = (const float*)d_in[2];
  const float* b1      = (const float*)d_in[3];
  const float* W2      = (const float*)d_in[4];
  const float* b2      = (const float*)d_in[5];
  const float* W3      = (const float*)d_in[6];
  const float* b3      = (const float*)d_in[7];
  const float* Jt      = (const float*)d_in[8];
  const float* sdirs   = (const float*)d_in[9];
  const float* hc      = (const float*)d_in[10];
  const float* hm      = (const float*)d_in[11];
  float* out           = (float*)d_out;
  bf16_t* wbf          = (bf16_t*)d_ws;

  // bf16 weight conversion (0.53 MB, L2-resident for the whole GEMM pass)
  convert_weights_kernel<<<(WS_TOT + 255) / 256, 256, 0, stream>>>(W1, W2, W3, wbf);

  hipFuncSetAttribute(reinterpret_cast<const void*>(hand_decoder_kernel),
                      hipFuncAttributeMaxDynamicSharedMemorySize, SMEM_BYTES);
  // 2048 blocks: [hand(2) x 1024 row-tiles of 64], 256 threads = 8 waves
  hand_decoder_kernel<<<2048, 256, SMEM_BYTES, stream>>>(
      z_left, z_right, b1, b2, b3, Jt, sdirs, hc, hm, wbf, out);
}
